// LTCSequence_816043786756
// MI455X (gfx1250) — matrix operand order repaired and checked by reading, not compiled, not run on hardware
//
#include <hip/hip_runtime.h>
#include <hip/hip_bf16.h>
#include <math.h>

typedef __attribute__((ext_vector_type(16))) _Float16 v16h;
typedef __attribute__((ext_vector_type(8)))  _Float16 v8h;
typedef __attribute__((ext_vector_type(8)))  float    v8f;
typedef __attribute__((ext_vector_type(4)))  float    v4f;

#define Bb   64
#define Ss   1024
#define INf  128
#define Hh   256
#define OUTf 128
#define DT_      0.1f
#define TAU_MIN_ 0.1f
#define TAU_MAX_ 10.0f

// f16 weight workspace layout (element offsets, halves):
//   TAU0 : 256 x 384   @ 0        (Wtau0, already [x|h] concatenated)
//   F0   : 256 x 384   @ 98304    ([Win0 | Wrec0])
//   TAU1 : 256 x 512   @ 196608   (Wtau1)
//   F1   : 256 x 512   @ 327680   ([Win1 | Wrec1])
//   WO   : 128 x 256   @ 458752   (Wout)
#define OFF_TAU0 0
#define OFF_F0   98304
#define OFF_TAU1 196608
#define OFF_F1   327680
#define OFF_WO   458752
#define WF_TOTAL 491520

union V8F { v8f v; float f[8]; };
union V4F { v4f v; float f[4]; };
union V8H { v8h v; _Float16 h[8]; };

static __device__ inline v8f splat8(float x) {
    V8F u;
#pragma unroll
    for (int i = 0; i < 8; ++i) u.f[i] = x;
    return u.v;
}

static __device__ inline float fast_rcp(float x) {
#if __has_builtin(__builtin_amdgcn_rcpf)
    return __builtin_amdgcn_rcpf(x);       // v_rcp_f32
#else
    return __fdividef(1.0f, x);
#endif
}

static __device__ inline float fast_tanh(float x) {
#if __has_builtin(__builtin_amdgcn_tanhf)
    return __builtin_amdgcn_tanhf(x);      // v_tanh_f32 (gfx1250 TRANS op)
#else
    float e = __expf(2.0f * x);            // tanh(x) = 1 - 2/(e^2x + 1)
    return 1.0f - 2.0f * fast_rcp(e + 1.0f);
#endif
}

static __device__ inline float fast_sigmoid(float x) {
    return fast_rcp(1.0f + __expf(-x));    // v_exp_f32 + v_rcp_f32
}

// Make a uniform pointer opaque to LICM so derived loads re-issue every loop
// iteration (streams the big layer-1 weight set from L2 instead of spilling).
static __device__ inline const _Float16* opaque_p(const _Float16* p) {
    asm volatile("" : "+s"(p));
    return p;
}

static __device__ inline v8f wmma_f16(v16h a, v16h b, v8f c) {
    // D = A(16x32 f16) * B(32x16 f16) + C(16x16 f32)
    return __builtin_amdgcn_wmma_f32_16x16x32_f16(
        /*neg_a=*/false, a, /*neg_b=*/false, b,
        /*c_mod=*/(short)0, c, /*reuse_a=*/false, /*reuse_b=*/false);
}

// A fragment (16x32 f16) from row-major LDS tile; lane = hi*16 + m.
// lanes 0-15: K = k0+0..15; lanes 16-31: K = k0+16..31.
static __device__ inline v16h load_a(const _Float16* lds, int stride, int m, int k0, int hi) {
    const _Float16* base = lds + m * stride + k0 + hi * 16;
    return *(const v16h*)(base);
}

// B fragment (32x16 f16) from row-major W[n][k]; lane's column n = tile_base + (lane&15),
// lanes 0-15 hold K = k0..k0+15, lanes 16-31 hold K = k0+16..k0+31 -> one 32B read per lane.
static __device__ inline v16h load_b(const _Float16* w, int stride, int n, int k0, int hi) {
    return *(const v16h*)(w + (size_t)n * stride + k0 + hi * 16);
}

// LTC state update on one 16x16 accumulator tile: col fixed per lane, rows via VGPR index.
static __device__ inline void update_h(_Float16* hbuf, int col, int hi, v8f tacc, v8f facc) {
    V8F ut, uf; ut.v = tacc; uf.v = facc;
#pragma unroll
    for (int i = 0; i < 8; ++i) {
        int row = i + hi * 8;
        float hold = (float)hbuf[row * Hh + col];
        float tau  = TAU_MIN_ + fast_sigmoid(ut.f[i]) * (TAU_MAX_ - TAU_MIN_);
        float fh   = fast_tanh(uf.f[i]);
        float rate = DT_ * fast_rcp(tau);
        hbuf[row * Hh + col] = (_Float16)(hold + rate * (fh - hold));
    }
}

// ---------------- weight fp32 -> f16 repack ----------------
__global__ void ltc_convert(const float* __restrict__ Win0, const float* __restrict__ Wrec0,
                            const float* __restrict__ Wtau0, const float* __restrict__ Win1,
                            const float* __restrict__ Wrec1, const float* __restrict__ Wtau1,
                            const float* __restrict__ Wout, _Float16* __restrict__ wf) {
    int idx = blockIdx.x * 256 + threadIdx.x;
    if (idx >= WF_TOTAL) return;
    float v;
    if (idx < OFF_F0) {
        v = Wtau0[idx];
    } else if (idx < OFF_TAU1) {
        int j = idx - OFF_F0; int n = j / 384; int k = j - n * 384;
        v = (k < INf) ? Win0[n * INf + k] : Wrec0[n * Hh + (k - INf)];
    } else if (idx < OFF_F1) {
        v = Wtau1[idx - OFF_TAU1];
    } else if (idx < OFF_WO) {
        int j = idx - OFF_F1; int n = j >> 9; int k = j & 511;
        v = (k < Hh) ? Win1[n * Hh + k] : Wrec1[n * Hh + (k - Hh)];
    } else {
        v = Wout[idx - OFF_WO];
    }
    wf[idx] = (_Float16)v;
}

// ---------------- persistent LTC sequence kernel ----------------
__global__ __launch_bounds__(256, 1) void ltc_seq(
    const float* __restrict__ x,
    const float* __restrict__ bin0, const float* __restrict__ brec0, const float* __restrict__ btau0,
    const float* __restrict__ bin1, const float* __restrict__ brec1, const float* __restrict__ btau1,
    const float* __restrict__ bout,
    const _Float16* __restrict__ wf,
    float* __restrict__ out) {
    __shared__ __align__(32) _Float16 xt[16 * INf];   // current x tile, f16
    __shared__ __align__(32) _Float16 h0s[16 * Hh];   // layer-0 state
    __shared__ __align__(32) _Float16 h1s[16 * Hh];   // layer-1 state

    const int tid  = threadIdx.x;
    const int wave = tid >> 5;
    const int lane = tid & 31;
    const int m    = lane & 15;
    const int hi   = lane >> 4;
    const int b0   = blockIdx.x * 16;     // batch-row tile base

    const _Float16* TAU0 = wf + OFF_TAU0;  // register-cacheable (layer 0 + Wout fit)
    const _Float16* F0   = wf + OFF_F0;
    const _Float16* WO   = wf + OFF_WO;

    const int n0 = wave * 32;   // this wave's two hidden N-tiles: n0, n0+16
    const int no = wave * 16;   // this wave's output N-tile

    // hoisted (t-invariant) biases
    const float bT0a = btau0[n0 + m],      bT0b = btau0[n0 + 16 + m];
    const float bF0a = bin0[n0 + m] + brec0[n0 + m];
    const float bF0b = bin0[n0 + 16 + m] + brec0[n0 + 16 + m];
    const float bT1a = btau1[n0 + m],      bT1b = btau1[n0 + 16 + m];
    const float bF1a = bin1[n0 + m] + brec1[n0 + m];
    const float bF1b = bin1[n0 + 16 + m] + brec1[n0 + 16 + m];
    const float bOo  = bout[no + m];

    for (int i = tid; i < 16 * Hh; i += 256) { h0s[i] = (_Float16)0.0f; h1s[i] = (_Float16)0.0f; }
    __syncthreads();

    for (int t = 0; t < Ss; ++t) {
        // layer-1 weights are streamed each step (too big to register-cache):
        const _Float16* TAU1 = opaque_p(wf + OFF_TAU1);
        const _Float16* F1   = opaque_p(wf + OFF_F1);

        // stage x[b0..b0+15, t, :] into LDS as f16 (vectorized: 8 floats/thread)
        {
            int e = tid * 8;                  // 2048 halves total
            int r = e >> 7, c = e & 127;
            const float* src = &x[((size_t)(b0 + r) * Ss + t) * INf + c];
            V4F x0, x1; x0.v = *(const v4f*)(src); x1.v = *(const v4f*)(src + 4);
            V8H hx;
#pragma unroll
            for (int i = 0; i < 4; ++i) { hx.h[i] = (_Float16)x0.f[i]; hx.h[4 + i] = (_Float16)x1.f[i]; }
            *(v8h*)(&xt[e]) = hx.v;
        }
        if (t + 1 < Ss && tid < 64) {   // prefetch next timestep tile (global_prefetch_b8)
            int r = tid >> 2, seg = tid & 3;
            __builtin_prefetch(&x[((size_t)(b0 + r) * Ss + (t + 1)) * INf + seg * 32], 0, 0);
        }
        __syncthreads();

        // ---------------- layer 0 ----------------
        {
            v8f tA = splat8(bT0a), tB = splat8(bT0b);
            v8f fA = splat8(bF0a), fB = splat8(bF0b);
#pragma unroll
            for (int k = 0; k < INf; k += 32) {          // x contribution
                v16h a = load_a(xt, INf, m, k, hi);
                tA = wmma_f16(a, load_b(TAU0, INf + Hh, n0 + m,      k, hi), tA);
                tB = wmma_f16(a, load_b(TAU0, INf + Hh, n0 + 16 + m, k, hi), tB);
                fA = wmma_f16(a, load_b(F0,   INf + Hh, n0 + m,      k, hi), fA);
                fB = wmma_f16(a, load_b(F0,   INf + Hh, n0 + 16 + m, k, hi), fB);
            }
#pragma unroll
            for (int k = 0; k < Hh; k += 32) {           // h0 contribution
                v16h a = load_a(h0s, Hh, m, k, hi);
                tA = wmma_f16(a, load_b(TAU0, INf + Hh, n0 + m,      INf + k, hi), tA);
                tB = wmma_f16(a, load_b(TAU0, INf + Hh, n0 + 16 + m, INf + k, hi), tB);
                fA = wmma_f16(a, load_b(F0,   INf + Hh, n0 + m,      INf + k, hi), fA);
                fB = wmma_f16(a, load_b(F0,   INf + Hh, n0 + 16 + m, INf + k, hi), fB);
            }
            __syncthreads();                              // all reads of h0s done
            update_h(h0s, n0 + m,      hi, tA, fA);
            update_h(h0s, n0 + 16 + m, hi, tB, fB);
        }
        __syncthreads();                                  // new h0 visible

        // ---------------- layer 1 (weights streamed from L2) ----------------
        {
            v8f tA = splat8(bT1a), tB = splat8(bT1b);
            v8f fA = splat8(bF1a), fB = splat8(bF1b);
#pragma unroll
            for (int k = 0; k < Hh; k += 32) {           // input = new h0
                v16h a = load_a(h0s, Hh, m, k, hi);
                tA = wmma_f16(a, load_b(TAU1, 2 * Hh, n0 + m,      k, hi), tA);
                tB = wmma_f16(a, load_b(TAU1, 2 * Hh, n0 + 16 + m, k, hi), tB);
                fA = wmma_f16(a, load_b(F1,   2 * Hh, n0 + m,      k, hi), fA);
                fB = wmma_f16(a, load_b(F1,   2 * Hh, n0 + 16 + m, k, hi), fB);
            }
#pragma unroll
            for (int k = 0; k < Hh; k += 32) {           // recurrent h1 (old)
                v16h a = load_a(h1s, Hh, m, k, hi);
                tA = wmma_f16(a, load_b(TAU1, 2 * Hh, n0 + m,      Hh + k, hi), tA);
                tB = wmma_f16(a, load_b(TAU1, 2 * Hh, n0 + 16 + m, Hh + k, hi), tB);
                fA = wmma_f16(a, load_b(F1,   2 * Hh, n0 + m,      Hh + k, hi), fA);
                fB = wmma_f16(a, load_b(F1,   2 * Hh, n0 + 16 + m, Hh + k, hi), fB);
            }
            __syncthreads();                              // all reads of h1s done
            update_h(h1s, n0 + m,      hi, tA, fA);
            update_h(h1s, n0 + 16 + m, hi, tB, fB);
        }
        __syncthreads();                                  // new h1 visible

        // ---------------- output projection ----------------
        {
            v8f o = splat8(bOo);
#pragma unroll
            for (int k = 0; k < Hh; k += 32) {
                v16h a = load_a(h1s, Hh, m, k, hi);
                o = wmma_f16(a, load_b(WO, Hh, no + m, k, hi), o);
            }
            V8F uo; uo.v = o;
#pragma unroll
            for (int i = 0; i < 8; ++i) {
                int row = i + hi * 8;
                out[((size_t)(b0 + row) * Ss + t) * OUTf + no + m] = uo.f[i];
            }
        }
        // next iter's xt writes can't race layer-0 reads (separated by barriers above)
    }

    __syncthreads();
    float* h0f = out + (size_t)Bb * Ss * OUTf;
    float* h1f = h0f + (size_t)Bb * Hh;
    for (int i = tid; i < 16 * Hh; i += 256) {
        int r = i >> 8, c = i & 255;
        h0f[(size_t)(b0 + r) * Hh + c] = (float)h0s[i];
        h1f[(size_t)(b0 + r) * Hh + c] = (float)h1s[i];
    }
}

extern "C" void kernel_launch(void* const* d_in, const int* in_sizes, int n_in,
                              void* d_out, int out_size, void* d_ws, size_t ws_size,
                              hipStream_t stream) {
    const float* x     = (const float*)d_in[0];
    const float* Win0  = (const float*)d_in[1];
    const float* bin0  = (const float*)d_in[2];
    const float* Wrec0 = (const float*)d_in[3];
    const float* brec0 = (const float*)d_in[4];
    const float* Wtau0 = (const float*)d_in[5];
    const float* btau0 = (const float*)d_in[6];
    const float* Win1  = (const float*)d_in[7];
    const float* bin1  = (const float*)d_in[8];
    const float* Wrec1 = (const float*)d_in[9];
    const float* brec1 = (const float*)d_in[10];
    const float* Wtau1 = (const float*)d_in[11];
    const float* btau1 = (const float*)d_in[12];
    const float* Wout  = (const float*)d_in[13];
    const float* bout  = (const float*)d_in[14];

    _Float16* wf = (_Float16*)d_ws;
    float* out = (float*)d_out;

    ltc_convert<<<(WF_TOTAL + 255) / 256, 256, 0, stream>>>(Win0, Wrec0, Wtau0,
                                                            Win1, Wrec1, Wtau1, Wout, wf);
    ltc_seq<<<Bb / 16, 256, 0, stream>>>(x, bin0, brec0, btau0,
                                         bin1, brec1, btau1, bout, wf, out);
}
